// ProductResidualLayer64_18124761989261
// MI455X (gfx1250) — matrix operand order repaired and checked by reading, not compiled, not run on hardware
//
#include <hip/hip_runtime.h>
#include <stdint.h>

// ---------------------------------------------------------------------------
// ProductResidualLayer64 on MI455X (gfx1250), compile-only tuned.
//
// out = prod @ w1 + x with prod[b,p] = x[b,I[p]]*x[b,J[p]] (i<j, P=2016).
// Symmetrized to a dense K=4096 GEMM: out = Xfull @ Wfull + x where
//   Xfull[b, i*64+j] = x[b,i]*x[b,j],  Wfull[(i,j),d] = 0.5*w1[pair(i,j),d]
//   (0 on the diagonal).  x_i*x_j = x_j*x_i makes this exactly equal.
// A-fragments are built with v_pk_mul_bf16 from register-resident bf16
// windows of the x row (no gathers); B is pre-repacked bf16 fragments
// streamed to double-buffered LDS via global_load_async_to_lds_b128, feeding
// v_wmma_f32_16x16x32_bf16 (f32 accumulate). Residual add in fp32.
// ---------------------------------------------------------------------------

typedef __attribute__((ext_vector_type(16))) __bf16    v16bf;
typedef __attribute__((ext_vector_type(8)))  float     v8f;
typedef __attribute__((ext_vector_type(8)))  uint32_t  v8u;

#define DIM       64
#define BATCH     131072
#define NCHUNK    128          // K = 4096 = 128 chunks of 32
#define CHUNK_DW  1024         // dwords of B-fragment data per chunk (32K x 64N bf16)
#define XI_STRIDE 65           // padded dword stride -> conflict-free xi broadcast

__device__ __forceinline__ uint32_t bf16_bits(float f) {
  uint32_t u = __float_as_uint(f);
  return (u + 0x7FFFu + ((u >> 16) & 1u)) >> 16;   // round-to-nearest-even
}
__device__ __forceinline__ uint32_t pack_bf16(float lo, float hi) {
  return bf16_bits(lo) | (bf16_bits(hi) << 16);
}

// --- pre-pass: repack w1 fp32[2016,64] -> Wfull bf16 WMMA-B fragments ------
// dword layout: bws[c*1024 + t*256 + lane*8 + v]
//   chunk c: i = c>>1, j0 = (c&1)*32
//   B frag (32x16 bf16): VGPR v<4 -> K=8h+2v,+1 ; v>=4 -> K=16+8h+2(v-4),+1
//                        (h = lane>>4); N = lane&15 (+16*t).
__global__ __launch_bounds__(256) void repack_w1_kernel(
    const float* __restrict__ w1, uint32_t* __restrict__ bws) {
  int idx  = blockIdx.x * 256 + threadIdx.x;   // 0 .. 131071
  int c    = idx >> 10;
  int rem  = idx & 1023;
  int t    = rem >> 8;
  int lane = (rem >> 3) & 31;
  int v    = rem & 7;
  int i    = c >> 1;
  int j0   = (c & 1) * 32;
  int kk   = (v >> 2) * 16 + (lane >> 4) * 8 + (v & 3) * 2;
  int col  = t * 16 + (lane & 15);
  int ja   = j0 + kk, jb = ja + 1;
  float fa = 0.f, fb = 0.f;
  if (ja != i) {
    int a = ja < i ? ja : i, b = ja < i ? i : ja;
    int p = a * 63 - (a * (a - 1)) / 2 + (b - a - 1);
    fa = 0.5f * w1[p * DIM + col];
  }
  if (jb != i) {
    int a = jb < i ? jb : i, b = jb < i ? i : jb;
    int p = a * 63 - (a * (a - 1)) / 2 + (b - a - 1);
    fb = 0.5f * w1[p * DIM + col];
  }
  bws[idx] = pack_bf16(fa, fb);
}

// --- main fused kernel -----------------------------------------------------
// 256 threads = 8 waves; each wave owns a 16-row tile (block = 128 rows) and
// the full N=64 as 4 WMMA accumulators.
__global__ __launch_bounds__(256) void prodres_wmma_kernel(
    const float* __restrict__ x, const uint32_t* __restrict__ bws,
    float* __restrict__ out) {
  __shared__ __align__(16) uint32_t lds_xi[128 * XI_STRIDE];  // 33.3 KB
  __shared__ __align__(32) uint32_t lds_b[2][CHUNK_DW];       //  8   KB

  const int tid     = threadIdx.x;
  const int lane    = tid & 31;
  const int waveId  = tid >> 5;
  const int m       = lane & 15;   // row within the wave's 16-row tile
  const int h       = lane >> 4;   // half-wave (A-fragment K-half)
  const int rowBase = blockIdx.x * 128;

  // Build xi broadcast table: duplicated bf16 pair of x[row][i], padded stride
  // so the hot-loop ds_load_b32 hits 16 distinct banks (2-lane broadcast each).
  {
    int rl   = tid >> 1;         // 0..127
    int half = tid & 1;
    const float* xr = x + (size_t)(rowBase + rl) * DIM + half * 32;
    uint32_t* dst = &lds_xi[rl * XI_STRIDE + half * 32];
#pragma unroll
    for (int k = 0; k < 32; k += 4) {
      float4 f = *(const float4*)(xr + k);
      dst[k + 0] = bf16_bits(f.x) * 0x10001u;
      dst[k + 1] = bf16_bits(f.y) * 0x10001u;
      dst[k + 2] = bf16_bits(f.z) * 0x10001u;
      dst[k + 3] = bf16_bits(f.w) * 0x10001u;
    }
  }

  // Per-lane A window: the 32 x-values this lane ever contributes, packed as
  // bf16 pairs already in WMMA-A fragment order.  16-bit A 16x32 layout:
  //   h=0: VGPR p<4 -> K=2p,2p+1 ; p>=4 -> K=16+2(p-4),...   (h=1: +8)
  // Chunk (i, jhalf): fragment = w[jhalf*8 + p] * dup(x[row][i]).
  const int    myRow = rowBase + waveId * 16 + m;
  const float* xrow  = x + (size_t)myRow * DIM;
  uint32_t w[16];
#pragma unroll
  for (int g = 0; g < 4; ++g) {
    const float* xp = xrow + g * 16 + 8 * h;
    float4 f0 = *(const float4*)(xp);
    float4 f1 = *(const float4*)(xp + 4);
    w[g * 4 + 0] = pack_bf16(f0.x, f0.y);
    w[g * 4 + 1] = pack_bf16(f0.z, f0.w);
    w[g * 4 + 2] = pack_bf16(f1.x, f1.y);
    w[g * 4 + 3] = pack_bf16(f1.z, f1.w);
  }

  v8f acc[4] = {};

  const uint64_t gbase  = (uint64_t)(uintptr_t)bws + (uint32_t)tid * 16u;
  const uint32_t lbase0 = ((uint32_t)(uintptr_t)&lds_b[0][0]) + (uint32_t)tid * 16u;
  const uint32_t lbase1 = ((uint32_t)(uintptr_t)&lds_b[1][0]) + (uint32_t)tid * 16u;

  __syncthreads();   // xi table ready

  // Prefetch chunk 0 into buffer 0 (each thread async-copies 16B -> 4KB/chunk)
  asm volatile("global_load_async_to_lds_b128 %0, %1, off"
               :: "v"(lbase0), "v"(gbase) : "memory");

  const int xiBase = (waveId * 16 + m) * XI_STRIDE;

  for (int i2 = 0; i2 < 64; ++i2) {
    const uint32_t xi = lds_xi[xiBase + i2];   // dup-bf16 x[row][i2]
#pragma unroll
    for (int jh = 0; jh < 2; ++jh) {
      const int c = i2 * 2 + jh;               // chunk; lives in lds_b[c&1]
      if (c + 1 < NCHUNK) {
        __syncthreads();  // all waves done reading the buffer we now overwrite
        uint64_t g = gbase + (uint64_t)(c + 1) * (CHUNK_DW * 4);
        uint32_t l = (jh == 0) ? lbase1 : lbase0;
        asm volatile("global_load_async_to_lds_b128 %0, %1, off"
                     :: "v"(l), "v"(g) : "memory");
        asm volatile("s_wait_asynccnt 0x1" ::: "memory");  // chunk c landed
      } else {
        asm volatile("s_wait_asynccnt 0x0" ::: "memory");
      }
      __syncthreads();    // chunk c visible to every wave

      // A fragment: x[row][i2] * x[row][j-window]  (8x v_pk_mul_bf16)
      v8u au;
#pragma unroll
      for (int p = 0; p < 8; ++p) {
        uint32_t r;
        asm("v_pk_mul_bf16 %0, %1, %2" : "=v"(r) : "v"(w[jh * 8 + p]), "v"(xi));
        au[p] = r;
      }
      v16bf af = __builtin_bit_cast(v16bf, au);

      // Issue all four B-fragment loads into independent register octets
      // first, so the WMMAs can run with staggered dscnt waits instead of a
      // full ds-latency stall per tile.
      v8u bu[4];
#pragma unroll
      for (int t = 0; t < 4; ++t)
        bu[t] = *(const v8u*)&lds_b[c & 1][t * 256 + lane * 8];

#pragma unroll
      for (int t = 0; t < 4; ++t) {
        v16bf bfrag = __builtin_bit_cast(v16bf, bu[t]);
        acc[t] = __builtin_amdgcn_wmma_f32_16x16x32_bf16(
            /*neg_a=*/false, af, /*neg_b=*/false, bfrag,
            /*c_mod=*/(short)0, acc[t], /*reuse_a=*/false, /*reuse_b=*/false);
      }
    }
  }

  // Epilogue: fp32 residual add + store.
  // C layout: N = lane&15 (+16t); M = v + 8*(lane>>4) -> per (t,v) a wave
  // writes two contiguous 64B row segments (coalesced).
#pragma unroll
  for (int t = 0; t < 4; ++t) {
    const int col = t * 16 + m;
#pragma unroll
    for (int v = 0; v < 8; ++v) {
      const int r = rowBase + waveId * 16 + v + 8 * h;
      const size_t off = (size_t)r * DIM + col;
      out[off] = acc[t][v] + x[off];
    }
  }
}

extern "C" void kernel_launch(void* const* d_in, const int* in_sizes, int n_in,
                              void* d_out, int out_size, void* d_ws, size_t ws_size,
                              hipStream_t stream) {
  (void)in_sizes; (void)n_in; (void)out_size; (void)ws_size;
  const float* x  = (const float*)d_in[0];   // [131072, 64] fp32
  const float* w1 = (const float*)d_in[1];   // [2016, 64]  fp32
  float* out = (float*)d_out;                // [131072, 64] fp32
  uint32_t* bws = (uint32_t*)d_ws;           // 512 KB repacked Wfull (bf16)

  repack_w1_kernel<<<NCHUNK * CHUNK_DW / 256, 256, 0, stream>>>(w1, bws);
  prodres_wmma_kernel<<<BATCH / 128, 256, 0, stream>>>(x, bws, out);
}
